// TriXLayer_5162550690202
// MI455X (gfx1250) — matrix operand order repaired and checked by reading, not compiled, not run on hardware
//
#include <hip/hip_runtime.h>
#include <hip/hip_bf16.h>
#include <stdint.h>

#define D_MODEL 1024
#define N_EXP   16
#define BATCH   32768
#define THRESH  0.3f

typedef __attribute__((ext_vector_type(16))) __bf16 v16bf;
typedef __attribute__((ext_vector_type(8)))  float  v8f;
typedef __attribute__((ext_vector_type(4)))  unsigned int v4u;
typedef __attribute__((ext_vector_type(8)))  int  v8i;
typedef __attribute__((ext_vector_type(4)))  int  v4i;

union BFrag { uint4 q[2]; v16bf v; };

// Build split-bf16 A fragment (16x32 bf16, contiguous k = 16*hh + s) for one wave.
__device__ __forceinline__ void load_a_split(const float* __restrict__ xr, int k0,
                                             v16bf& ah, v16bf& al) {
  float4 a0 = *(const float4*)(xr + k0);
  float4 a1 = *(const float4*)(xr + k0 + 4);
  float4 a2 = *(const float4*)(xr + k0 + 8);
  float4 a3 = *(const float4*)(xr + k0 + 12);
  float fa[16] = {a0.x,a0.y,a0.z,a0.w, a1.x,a1.y,a1.z,a1.w,
                  a2.x,a2.y,a2.z,a2.w, a3.x,a3.y,a3.z,a3.w};
#pragma unroll
  for (int i = 0; i < 16; i++) {
    __bf16 h = (__bf16)fa[i];
    ah[i] = h;
    al[i] = (__bf16)(fa[i] - (float)h);
  }
}

__device__ __forceinline__ uint32_t pack_bf16(__bf16 a, __bf16 b) {
  return (uint32_t)__builtin_bit_cast(unsigned short, a) |
         ((uint32_t)__builtin_bit_cast(unsigned short, b) << 16);
}

// --- ASYNCcnt path (4B-granular; used for the compacted row-id tile) ---
__device__ __forceinline__ void async_b32(uint32_t lds_off, const void* gaddr) {
  asm volatile("global_load_async_to_lds_b32 %0, %1, off"
               :: "v"(lds_off), "v"(gaddr) : "memory");
}
__device__ __forceinline__ void wait_async0() {
  asm volatile("s_wait_asynccnt 0" ::: "memory");
}

// --- TENSORcnt path: TDM 2D tile load, 128 rows x 64B, bf16 elements,
// LDS padded +16B every 64B (row stride 80B). 6-arg clang-23 builtin form. ---
#define LROW 40  // LDS row stride in bf16 elems (80 B)

__device__ __forceinline__ void tdm_load_tile(const __bf16* gsrc, uint32_t lds_off) {
  uint64_t ga = (uint64_t)(uintptr_t)gsrc;
  v4u g0;
  g0.x = 1u;                                   // count=1, user descriptor
  g0.y = lds_off;                              // lds_addr [63:32]
  g0.z = (uint32_t)ga;                         // global_addr lo
  g0.w = (uint32_t)(ga >> 32) | (2u << 30);    // global_addr hi | type=2
  v8i g1;
  g1[0] = (int)((1u << 16)     // data_size = 1 -> 2 bytes
              | (1u << 20)     // pad_enable
              | (3u << 22)     // pad_interval = 3 -> every 16 DWORDs (64B)
              | (3u << 25));   // pad_amount   = 3 -> 4 DWORDs (16B)
  g1[1] = (int)(1024u << 16);  // tensor_dim0 = 1024 (bits [79:48], lo16 here)
  g1[2] = (int)(1024u << 16);  // tensor_dim1 = 1024 (bits [111:80], lo16 here)
  g1[3] = (int)(32u << 16);    // tile_dim0 = 32 elems (64B) at [127:112]
  g1[4] = 128;                 // tile_dim1 = 128 rows; tile_dim2 = 0
  g1[5] = 1024;                // tensor_dim0_stride = 1024 elems (lo32)
  g1[6] = 0;
  g1[7] = 0;
  v4i z4 = {0, 0, 0, 0};            // groups 2/3 unused (2D tile)
  v8i z8 = {0, 0, 0, 0, 0, 0, 0, 0};
  __builtin_amdgcn_tensor_load_to_lds(g0, g1, z4, z4, z8, 0);
}

// ---------------- kernel 0: split W into bf16 hi/lo (once per call)
__global__ __launch_bounds__(256) void wsplit_kernel(const float* __restrict__ W,
                                                     __bf16* __restrict__ Wh,
                                                     __bf16* __restrict__ Wl) {
  size_t i = ((size_t)blockIdx.x * blockDim.x + threadIdx.x) * 4;
  float4 wv = *(const float4*)(W + i);
  float f[4] = {wv.x, wv.y, wv.z, wv.w};
  __bf16 h[4], l[4];
#pragma unroll
  for (int j = 0; j < 4; j++) {
    h[j] = (__bf16)f[j];
    l[j] = (__bf16)(f[j] - (float)h[j]);
  }
  uint2 ph = {pack_bf16(h[0], h[1]), pack_bf16(h[2], h[3])};
  uint2 pl = {pack_bf16(l[0], l[1]), pack_bf16(l[2], l[3])};
  *(uint2*)(Wh + i) = ph;
  *(uint2*)(Wl + i) = pl;
}

// ---------------- kernel 1: quantize signatures, zero counts
__global__ void prep_kernel(const float* __restrict__ raw, __bf16* __restrict__ sig,
                            int* __restrict__ counts) {
  int i = blockIdx.x * blockDim.x + threadIdx.x;
  if (i < N_EXP) counts[i] = 0;
  if (i < N_EXP * D_MODEL) {
    float s = raw[i];
    float q = (s > THRESH) ? 1.0f : ((s < -THRESH) ? -1.0f : 0.0f);
    sig[i] = (__bf16)q;
  }
}

// ---------------- kernel 2: WMMA routing scores + argmax + histogram
__global__ __launch_bounds__(128) void route_kernel(const float* __restrict__ x,
                                                    const __bf16* __restrict__ sig,
                                                    int* __restrict__ tidx,
                                                    int* __restrict__ counts) {
  int lane = threadIdx.x & 31, w = threadIdx.x >> 5;
  int hh = lane >> 4, lm = lane & 15;
  int rb = blockIdx.x * 64 + w * 16;

  const float*  xr = x + (size_t)(rb + lm) * D_MODEL;
  const __bf16* sp = sig + (size_t)lm * D_MODEL + hh * 16;

  v8f c = {};
  for (int kb = 0; kb < D_MODEL; kb += 32) {
    v16bf ah, al;
    load_a_split(xr, kb + hh * 16, ah, al);
    BFrag s;
    s.q[0] = *(const uint4*)(sp + kb);
    s.q[1] = *(const uint4*)(sp + kb + 8);
    c = __builtin_amdgcn_wmma_f32_16x16x32_bf16(false, ah, false, s.v, (short)0, c, false, false);
    c = __builtin_amdgcn_wmma_f32_16x16x32_bf16(false, al, false, s.v, (short)0, c, false, false);
  }

#pragma unroll
  for (int r = 0; r < 8; r++) {
    float sc = c[r];
    int   bi = lm;
#pragma unroll
    for (int off = 1; off < 16; off <<= 1) {
      float os = __shfl_xor(sc, off, 32);
      int   oi = __shfl_xor(bi, off, 32);
      if (os > sc || (os == sc && oi < bi)) { sc = os; bi = oi; }
    }
    if (lm == 0) {
      int row = rb + r + 8 * hh;
      tidx[row] = bi;
      atomicAdd(&counts[bi], 1);
    }
  }
}

// ---------------- kernel 3: tiny exclusive scan
__global__ void scan_kernel(const int* __restrict__ counts, int* __restrict__ offsets,
                            int* __restrict__ cursors) {
  if (threadIdx.x == 0) {
    int a = 0;
    for (int e = 0; e < N_EXP; e++) { offsets[e] = a; cursors[e] = 0; a += counts[e]; }
  }
}

// ---------------- kernel 4: compact row ids by expert
__global__ void gather_kernel(const int* __restrict__ tidx, int* __restrict__ cursors,
                              const int* __restrict__ offsets, int* __restrict__ rowids) {
  int r = blockIdx.x * blockDim.x + threadIdx.x;
  if (r < BATCH) {
    int e = tidx[r];
    int p = atomicAdd(&cursors[e], 1);
    rowids[offsets[e] + p] = r;
  }
}

// ---------------- kernel 5: grouped GEMM, M64 x N128 per block (256 thr, 8 waves).
// W k-slices (pre-split bf16 hi/lo) streamed into double-buffered LDS by the
// Tensor Data Mover; row-id tile via async-to-LDS; 12 WMMAs/wave/k-step.
__global__ __launch_bounds__(256) void gemm_kernel(const float* __restrict__ x,
                                                   const __bf16* __restrict__ Wh,
                                                   const __bf16* __restrict__ Wl,
                                                   const float* __restrict__ bias,
                                                   const int* __restrict__ rowids,
                                                   const int* __restrict__ counts,
                                                   const int* __restrict__ offsets,
                                                   float* __restrict__ out) {
  __shared__ __align__(16) __bf16 ldsWh[2][128 * LROW];
  __shared__ __align__(16) __bf16 ldsWl[2][128 * LROW];
  __shared__ __align__(16) int srow[64];

  // Map blockIdx.x -> (expert, m-tile).
  int bid = blockIdx.x;
  int e = -1, trow = 0, cnt = 0, off = 0, acct = 0;
#pragma unroll
  for (int i = 0; i < N_EXP; i++) {
    int ci = counts[i];
    int nt = (ci + 63) >> 6;
    if (e < 0 && bid < acct + nt) { e = i; trow = (bid - acct) << 6; cnt = ci; off = offsets[i]; }
    acct += nt;
  }
  if (e < 0) return;  // uniform across block

  int tid = threadIdx.x;
  // Row-id tile -> LDS via ASYNCcnt path (4B aligned source; rowids has slack).
  if (tid < 16) {
#pragma unroll
    for (int c = 0; c < 4; c++) {
      int m = tid * 4 + c;
      async_b32((uint32_t)(uintptr_t)&srow[m], (const void*)&rowids[off + trow + m]);
    }
  }

  int lane = tid & 31, w = tid >> 5;
  int hh = lane >> 4, lm = lane & 15;
  int mg = w & 3;        // m-group (16 rows) -> M64
  int ng = w >> 2;       // n-group (64 cols) -> N128

  const int nbase = blockIdx.y * 128;
  const __bf16* Whe = Wh + ((size_t)e << 20) + (size_t)nbase * D_MODEL;
  const __bf16* Wle = Wl + ((size_t)e << 20) + (size_t)nbase * D_MODEL;

  // Prologue: TDM-stream k-slice 0 into buffer 0 (wave 0 drives the DMA).
  if (w == 0) {
    tdm_load_tile(Whe, (uint32_t)(uintptr_t)&ldsWh[0][0]);
    tdm_load_tile(Wle, (uint32_t)(uintptr_t)&ldsWl[0][0]);
  }
  wait_async0();
  __syncthreads();  // srow visible

  bool va = (trow + mg * 16 + lm) < cnt;
  int  gac = va ? srow[mg * 16 + lm] : srow[0];  // clamp: WMMA needs full EXEC
  const float* xr = x + (size_t)gac * D_MODEL;

  v8f acc[4] = {};

  for (int it = 0; it < D_MODEL / 32; ++it) {
    int kb  = it * 32;
    int cur = it & 1;

    if (w == 0) __builtin_amdgcn_s_wait_tensorcnt(0);
    __syncthreads();  // buf[cur] ready; buf[1-cur] no longer being read

    if (it + 1 < D_MODEL / 32) {
      if (w == 0) {
        tdm_load_tile(Whe + kb + 32, (uint32_t)(uintptr_t)&ldsWh[1 - cur][0]);
        tdm_load_tile(Wle + kb + 32, (uint32_t)(uintptr_t)&ldsWl[1 - cur][0]);
      }
      __builtin_prefetch(xr + kb + 32, 0, 3);  // next x k-slice -> global_prefetch
    }

    v16bf ah, al;
    load_a_split(xr, kb + hh * 16, ah, al);

#pragma unroll
    for (int nt = 0; nt < 4; nt++) {
      int n = ng * 64 + nt * 16 + lm;
      BFrag bh, bl;
      const __bf16* ph = &ldsWh[cur][n * LROW + hh * 16];
      const __bf16* pl = &ldsWl[cur][n * LROW + hh * 16];
      bh.q[0] = *(const uint4*)(ph);
      bh.q[1] = *(const uint4*)(ph + 8);
      bl.q[0] = *(const uint4*)(pl);
      bl.q[1] = *(const uint4*)(pl + 8);
      acc[nt] = __builtin_amdgcn_wmma_f32_16x16x32_bf16(false, ah, false, bh.v, (short)0, acc[nt], false, false);
      acc[nt] = __builtin_amdgcn_wmma_f32_16x16x32_bf16(false, ah, false, bl.v, (short)0, acc[nt], false, false);
      acc[nt] = __builtin_amdgcn_wmma_f32_16x16x32_bf16(false, al, false, bh.v, (short)0, acc[nt], false, false);
    }
  }

  // Epilogue: bias + ReLU + residual.
#pragma unroll
  for (int nt = 0; nt < 4; nt++) {
    int   n  = nbase + ng * 64 + nt * 16 + lm;
    float bv = bias[e * D_MODEL + n];
#pragma unroll
    for (int r = 0; r < 8; r++) {
      int m = mg * 16 + r + 8 * hh;  // C layout: VGPR r -> M = r + 8*hh
      if (trow + m < cnt) {
        int gc = srow[m];
        float v = acc[nt][r] + bv;
        v = v > 0.0f ? v : 0.0f;
        v += x[(size_t)gc * D_MODEL + n];
        out[(size_t)gc * D_MODEL + n] = v;
      }
    }
  }
}

// ---------------- launcher
extern "C" void kernel_launch(void* const* d_in, const int* in_sizes, int n_in,
                              void* d_out, int out_size, void* d_ws, size_t ws_size,
                              hipStream_t stream) {
  const float* x    = (const float*)d_in[0];
  const float* sraw = (const float*)d_in[1];
  const float* W    = (const float*)d_in[2];
  const float* b    = (const float*)d_in[3];

  float* out  = (float*)d_out;
  int*   tidx = (int*)(out + (size_t)BATCH * D_MODEL);

  char*   ws      = (char*)d_ws;
  __bf16* Wh      = (__bf16*)ws;                          // 32 MB
  __bf16* Wl      = (__bf16*)(ws + ((size_t)32 << 20));   // 32 MB
  char*   ws2     = ws + ((size_t)64 << 20);
  __bf16* sig_bf  = (__bf16*)ws2;                         // 32 KB
  int*    counts  = (int*)(ws2 + 32768);
  int*    offsets = (int*)(ws2 + 32768 + 64);
  int*    cursors = (int*)(ws2 + 32768 + 128);
  int*    rowids  = (int*)(ws2 + 32768 + 192);            // 128 KB + 64-int slack

  wsplit_kernel<<<(N_EXP * D_MODEL * D_MODEL) / (256 * 4), 256, 0, stream>>>(W, Wh, Wl);
  prep_kernel<<<64, 256, 0, stream>>>(sraw, sig_bf, counts);
  route_kernel<<<BATCH / 64, 128, 0, stream>>>(x, sig_bf, tidx, counts);
  scan_kernel<<<1, 32, 0, stream>>>(counts, offsets, cursors);
  gather_kernel<<<BATCH / 256, 256, 0, stream>>>(tidx, cursors, offsets, rowids);

  dim3 grid(BATCH / 64 + N_EXP, D_MODEL / 128);
  gemm_kernel<<<grid, 256, 0, stream>>>(x, Wh, Wl, b, rowids, counts, offsets, out);
}
